// MultiHeadAttention_62405874811119
// MI455X (gfx1250) — matrix operand order repaired and checked by reading, not compiled, not run on hardware
//
#include <hip/hip_runtime.h>

// ---------------------------------------------------------------------------
// MultiHeadAttention for MI455X (gfx1250, wave32, WMMA + async-LDS copies).
// B=2, S=2048, D=1024, H=16, DK=64.  fp32 in/out, f16 WMMA with f32 accum.
// ---------------------------------------------------------------------------

typedef __attribute__((ext_vector_type(16))) _Float16 v16h;
typedef __attribute__((ext_vector_type(8)))  float    v8f;
typedef __attribute__((ext_vector_type(4)))  _Float16 h4;
typedef __attribute__((ext_vector_type(4)))  int      vi4;

#if __has_builtin(__builtin_amdgcn_global_load_async_to_lds_b128)
#define ASYNC_LDS 1
#endif

// Copy one 16-byte chunk global -> LDS (async DMA path when available).
__device__ __forceinline__ void cp16(_Float16* l, const _Float16* g) {
#ifdef ASYNC_LDS
  __builtin_amdgcn_global_load_async_to_lds_b128((vi4*)g, (vi4*)l, 0, 0);
#else
  *(uint4*)l = *(const uint4*)g;
#endif
}

__device__ __forceinline__ void wait_async_all() {
#ifdef ASYNC_LDS
  asm volatile("s_wait_asynccnt 0x0" ::: "memory");
#endif
}

__device__ __forceinline__ v8f wmma_f16(v16h a, v16h b, v8f c) {
  // D(16x16,f32) = A(16x32,f16) x B(32x16,f16) + C
  return __builtin_amdgcn_wmma_f32_16x16x32_f16(false, a, false, b,
                                                (short)0, c, false, false);
}

// A-fragment (16x32 f16, row-major M x K). lane&15 = M row, lane>>4 = K-half;
// VGPRs 0-3 -> K = khi*8+0..7, VGPRs 4-7 -> +16.
__device__ __forceinline__ v16h ld_afrag(const _Float16* p0, int rowStride) {
  const int lane = threadIdx.x & 31;
  const _Float16* p = p0 + (lane & 15) * rowStride + ((lane >> 4) << 3);
  v16h f;
  ((uint4*)&f)[0] = *(const uint4*)(p);
  ((uint4*)&f)[1] = *(const uint4*)(p + 16);
  return f;
}

// B-fragment (32x16 f16, row-major K x N). lane&15 = N col, lane>>4 = K-half;
// elements 0-7 -> K = khi*8+0..7, elements 8-15 -> +16.
__device__ __forceinline__ v16h ld_bfrag(const _Float16* p0, int rowStride) {
  const int lane = threadIdx.x & 31;
  const _Float16* p = p0 + (lane & 15) + ((lane >> 4) << 3) * rowStride;
  v16h f;
  for (int s = 0; s < 8; ++s) {
    f[s] = p[s * rowStride];
    f[8 + s] = p[(16 + s) * rowStride];
  }
  return f;
}

// ---------------------------------------------------------------------------
// Pre-passes: fp32 -> f16 activations; fp32 W -> f16 W^T (K x N).
// ---------------------------------------------------------------------------
__global__ __launch_bounds__(256) void cvt_f32_f16(const float* __restrict__ x,
                                                   _Float16* __restrict__ y) {
  const int i = (blockIdx.x * 256 + threadIdx.x) * 4;
  const float4 v = *(const float4*)(x + i);
  h4 hv = {(_Float16)v.x, (_Float16)v.y, (_Float16)v.z, (_Float16)v.w};
  *(h4*)(y + i) = hv;
}

__global__ __launch_bounds__(256) void transpose_wT(const float* __restrict__ W,
                                                    _Float16* __restrict__ Wt) {
  __shared__ float t[32][33];
  const int n0 = blockIdx.x * 32;             // row block of W (output feature)
  const int k0 = blockIdx.y * 32;             // col block of W (input feature)
  const int tx = threadIdx.x & 31, ty = threadIdx.x >> 5;  // 32 x 8
  for (int r = 0; r < 32; r += 8)
    t[ty + r][tx] = W[(size_t)(n0 + ty + r) * 1024 + k0 + tx];
  __syncthreads();
  for (int r = 0; r < 32; r += 8)
    Wt[(size_t)(k0 + ty + r) * 1024 + n0 + tx] = (_Float16)t[tx][ty + r];
}

// ---------------------------------------------------------------------------
// GEMM: Y[m,n] = sum_k A[m,k] * Bt[k,n]   (M=4096, N=1024, K=1024, f16 in).
// OUT_HEADS=1: f16 head layout (B,H,S,DK) with scale. OUT_HEADS=0: fp32 row-major.
// 128x128 C-tile, BK=32, 8 waves (2Mx4N), wave tile 64x32.
// Double-buffered LDS fed by async global->LDS b128 copies.
// ---------------------------------------------------------------------------
template <int OUT_HEADS>
__global__ __launch_bounds__(256) void gemm_f16(const _Float16* __restrict__ A,
                                                const _Float16* __restrict__ Bt,
                                                _Float16* __restrict__ Yh,
                                                float* __restrict__ Yf,
                                                float scale) {
  constexpr int K = 1024, N = 1024;
  constexpr int LDA = 40;   // 80B rows: conflict-free b128 fragment reads
  constexpr int LDB = 136;  // 272B rows
  __shared__ __align__(16) _Float16 lA[2][128 * LDA];   // [m][k]
  __shared__ __align__(16) _Float16 lB[2][32 * LDB];    // [k][n]

  const int m0 = blockIdx.x * 128;
  const int n0 = blockIdx.y * 128;
  const int tid = threadIdx.x;
  const int w = tid >> 5;
  const int waveM = (w & 1) * 64;
  const int waveN = (w >> 1) * 32;

  auto stage = [&](int buf, int k0) {
    // A tile 128x32 f16: 512 x 16B chunks, 2 per thread.
    for (int p = 0; p < 2; ++p) {
      const int q = p * 256 + tid;
      const int r = q >> 2, c = (q & 3) << 3;
      cp16(&lA[buf][r * LDA + c], A + (size_t)(m0 + r) * K + k0 + c);
    }
    // Bt tile 32x128 f16: 512 x 16B chunks, 2 per thread.
    for (int p = 0; p < 2; ++p) {
      const int q = p * 256 + tid;
      const int r = q >> 4, c = (q & 15) << 3;
      cp16(&lB[buf][r * LDB + c], Bt + (size_t)(k0 + r) * N + n0 + c);
    }
  };

  v8f acc[4][2] = {};
  stage(0, 0);
  for (int k0 = 0, it = 0; k0 < K; k0 += 32, ++it) {
    wait_async_all();
    __syncthreads();                      // tile `it` visible to all waves
    if (k0 + 32 < K) stage((it + 1) & 1, k0 + 32);  // overlap next-tile DMA
    const _Float16* cA = lA[it & 1];
    const _Float16* cB = lB[it & 1];
    v16h a[4], b[2];
    for (int i = 0; i < 4; ++i) a[i] = ld_afrag(&cA[(waveM + i * 16) * LDA], LDA);
    for (int j = 0; j < 2; ++j) b[j] = ld_bfrag(&cB[waveN + j * 16], LDB);
    for (int i = 0; i < 4; ++i)
      for (int j = 0; j < 2; ++j) acc[i][j] = wmma_f16(a[i], b[j], acc[i][j]);
    __syncthreads();                      // all reads of this buffer done
  }

  // C/D layout: VGPR i holds M = i + 8*(lane>=16), N = lane&15.
  const int lane = tid & 31;
  const int colL = lane & 15;
  const int rhi = (lane >> 4) << 3;
  for (int i4 = 0; i4 < 4; ++i4)
    for (int j = 0; j < 2; ++j)
      for (int i = 0; i < 8; ++i) {
        const int m = m0 + waveM + i4 * 16 + i + rhi;
        const int n = n0 + waveN + j * 16 + colL;
        const float val = acc[i4][j][i];
        if (OUT_HEADS) {
          const int bb = m >> 11, ss = m & 2047;   // S=2048
          const int hh = n >> 6, dk = n & 63;      // DK=64
          Yh[(((size_t)(bb * 16 + hh)) * 2048 + ss) * 64 + dk] =
              (_Float16)(val * scale);
        } else {
          Yf[(size_t)m * 1024 + n] = val;
        }
      }
}

// ---------------------------------------------------------------------------
// Causal flash attention. Grid: (B*H) * (S/128). 8 waves; wave = 16 q-rows.
// Q pre-scaled by 1/sqrt(DK). 32-key tiles; V staged via async DMA; online
// softmax in C-fragment register layout; concat written as f16.
// ---------------------------------------------------------------------------
__global__ __launch_bounds__(256) void flash_attn(const _Float16* __restrict__ Qh,
                                                  const _Float16* __restrict__ Kh,
                                                  const _Float16* __restrict__ Vh,
                                                  _Float16* __restrict__ Ch) {
  constexpr int LDK = 40, LDV = 72, LDP = 40;
  __shared__ __align__(16) _Float16 lKt[64 * LDK];    // [dk][key] (transposed)
  __shared__ __align__(16) _Float16 lV[32 * LDV];     // [key][dk]
  __shared__ __align__(16) _Float16 lP[8 * 16 * LDP]; // per-wave P scratch

  const int tid = threadIdx.x;
  const int w = tid >> 5;
  const int lane = tid & 31;
  const int qb = blockIdx.x & 15;          // S/128 = 16 query blocks
  const int bh = blockIdx.x >> 4;          // b*H + h
  const size_t base = (size_t)bh * 2048 * 64;
  const int qr0 = qb * 128 + w * 16;
  const int colL = lane & 15;
  const int rhi = (lane >> 4) << 3;

  v16h qa[2];
  qa[0] = ld_afrag(Qh + base + (size_t)qr0 * 64, 64);
  qa[1] = ld_afrag(Qh + base + (size_t)qr0 * 64 + 32, 64);

  v8f o[4] = {};
  float mr[8], lr[8];
  for (int i = 0; i < 8; ++i) { mr[i] = -3.0e38f; lr[i] = 0.f; }

  _Float16* lPw = &lP[w * 16 * LDP];
  const int kend = qb * 128 + 128;
  const int key = tid >> 3;
  const int dk0 = (tid & 7) << 3;

  for (int kt = 0; kt < kend; kt += 32) {
    {  // stage K (transpose via VGPRs) + V (straight async DMA), 32x64 each
      const _Float16* kg = Kh + base + (size_t)(kt + key) * 64 + dk0;
      const _Float16* vg = Vh + base + (size_t)(kt + key) * 64 + dk0;
      cp16(&lV[key * LDV + dk0], vg);
      const uint4 kv = *(const uint4*)kg;
      const _Float16* kh = (const _Float16*)&kv;
      for (int j = 0; j < 8; ++j) lKt[(dk0 + j) * LDK + key] = kh[j];
      if (kt + 32 < kend) {  // pull next tile toward L2/WGP$
        __builtin_prefetch(kg + 32 * 64, 0, 1);
        __builtin_prefetch(vg + 32 * 64, 0, 1);
      }
    }
    wait_async_all();
    __syncthreads();

    if (kt <= qr0 + 15) {  // skip tiles entirely above this wave's diagonal
      v8f sc[2] = {};
      for (int st = 0; st < 2; ++st) {
        sc[st] = wmma_f16(qa[0], ld_bfrag(&lKt[st * 16], LDK), sc[st]);
        sc[st] = wmma_f16(qa[1], ld_bfrag(&lKt[32 * LDK + st * 16], LDK), sc[st]);
      }
      // causal mask + online softmax (rows live in 16-lane halves)
      float al[8];
      for (int i = 0; i < 8; ++i) {
        const int qrow = qr0 + i + rhi;
        if (kt + colL > qrow)      sc[0][i] = -3.0e38f;
        if (kt + 16 + colL > qrow) sc[1][i] = -3.0e38f;
        float v = fmaxf(sc[0][i], sc[1][i]);
        for (int d = 8; d >= 1; d >>= 1) v = fmaxf(v, __shfl_xor(v, d, 32));
        const float mn = fmaxf(mr[i], v);
        al[i] = __expf(mr[i] - mn);
        mr[i] = mn;
      }
      for (int i = 0; i < 8; ++i) {
        sc[0][i] = __expf(sc[0][i] - mr[i]);
        sc[1][i] = __expf(sc[1][i] - mr[i]);
        float v = sc[0][i] + sc[1][i];
        for (int d = 8; d >= 1; d >>= 1) v += __shfl_xor(v, d, 32);
        lr[i] = lr[i] * al[i] + v;
      }
      for (int n = 0; n < 4; ++n)
        for (int i = 0; i < 8; ++i) o[n][i] *= al[i];
      // C-frag (f32) -> A-frag (f16) re-layout through per-wave LDS
      for (int st = 0; st < 2; ++st)
        for (int i = 0; i < 8; ++i)
          lPw[(i + rhi) * LDP + st * 16 + colL] = (_Float16)sc[st][i];
      asm volatile("s_wait_dscnt 0x0" ::: "memory");  // in-wave LDS RAW order
      const v16h pa = ld_afrag(lPw, LDP);
      for (int n = 0; n < 4; ++n)
        o[n] = wmma_f16(pa, ld_bfrag(&lV[n * 16], LDV), o[n]);
    }
    __syncthreads();
  }

  // normalize; write concat (B,S,D) as f16, column h*64+dk
  const int b = bh >> 4, h = bh & 15;
  for (int n = 0; n < 4; ++n)
    for (int i = 0; i < 8; ++i) {
      const int q = qr0 + i + rhi;
      Ch[((size_t)(b * 2048 + q)) * 1024 + h * 64 + n * 16 + colL] =
          (_Float16)(o[n][i] * (1.0f / lr[i]));
    }
}

// ---------------------------------------------------------------------------
extern "C" void kernel_launch(void* const* d_in, const int* in_sizes, int n_in,
                              void* d_out, int out_size, void* d_ws, size_t ws_size,
                              hipStream_t stream) {
  const float* q  = (const float*)d_in[0];
  const float* k  = (const float*)d_in[1];
  const float* v  = (const float*)d_in[2];
  // d_in[3] = causal mask: structure known (tril), applied analytically.
  const float* Wq = (const float*)d_in[4];
  const float* Wk = (const float*)d_in[5];
  const float* Wv = (const float*)d_in[6];
  const float* Wo = (const float*)d_in[7];

  char* ws = (char*)d_ws;
  const size_t MB = 1024 * 1024;
  _Float16* Xq  = (_Float16*)(ws + 0 * MB);   // 8 MB  f16 activations
  _Float16* Xk  = (_Float16*)(ws + 8 * MB);
  _Float16* Xv  = (_Float16*)(ws + 16 * MB);
  _Float16* Wqt = (_Float16*)(ws + 24 * MB);  // 2 MB each, f16 W^T (KxN)
  _Float16* Wkt = (_Float16*)(ws + 26 * MB);
  _Float16* Wvt = (_Float16*)(ws + 28 * MB);
  _Float16* Wot = (_Float16*)(ws + 30 * MB);
  _Float16* Qh  = (_Float16*)(ws + 32 * MB);  // 8 MB each, head layout
  _Float16* Kh  = (_Float16*)(ws + 40 * MB);
  _Float16* Vh  = (_Float16*)(ws + 48 * MB);
  _Float16* Chc = (_Float16*)(ws + 56 * MB);  // 8 MB concat f16

  const dim3 blk(256);
  const dim3 gCvt(4096);       // 4M elems / (256*4)
  const dim3 gTr(32, 32);      // 1024/32 x 1024/32
  const dim3 gGemm(32, 8);     // M/128 x N/128

  cvt_f32_f16<<<gCvt, blk, 0, stream>>>(q, Xq);
  cvt_f32_f16<<<gCvt, blk, 0, stream>>>(k, Xk);
  cvt_f32_f16<<<gCvt, blk, 0, stream>>>(v, Xv);
  transpose_wT<<<gTr, blk, 0, stream>>>(Wq, Wqt);
  transpose_wT<<<gTr, blk, 0, stream>>>(Wk, Wkt);
  transpose_wT<<<gTr, blk, 0, stream>>>(Wv, Wvt);
  transpose_wT<<<gTr, blk, 0, stream>>>(Wo, Wot);

  gemm_f16<1><<<gGemm, blk, 0, stream>>>(Xq, Wqt, Qh, nullptr, 0.125f); // 1/sqrt(64)
  gemm_f16<1><<<gGemm, blk, 0, stream>>>(Xk, Wkt, Kh, nullptr, 1.0f);
  gemm_f16<1><<<gGemm, blk, 0, stream>>>(Xv, Wvt, Vh, nullptr, 1.0f);

  flash_attn<<<dim3(2 * 16 * 16), blk, 0, stream>>>(Qh, Kh, Vh, Chc);

  gemm_f16<0><<<gGemm, blk, 0, stream>>>(Chc, Wot, nullptr, (float*)d_out, 1.0f);
}
